// MLPWithLayerNormSiLU_73787538145416
// MI455X (gfx1250) — matrix operand order repaired and checked by reading, not compiled, not run on hardware
//
#include <hip/hip_runtime.h>
#include <cmath>

typedef _Float16 half_t;
typedef __attribute__((ext_vector_type(16))) _Float16 v16h;
typedef __attribute__((ext_vector_type(8)))  float    v8f;

#define EPSV 1e-5f

__device__ __forceinline__ void lds_fence() {
    // cross-lane LDS RAW inside a wave: wait for DS ops to complete
    asm volatile("s_wait_dscnt 0" ::: "memory");
}

// CDNA5 has a hardware tanh transcendental (TANH listed in the trans-op set).
// silu(x) = x*sigmoid(x) = s*(1+tanh(s)) with s = x/2 -> ONE trans op instead
// of exp+rcp (two). Trans throughput is the dominant non-WMMA cost here.
__device__ __forceinline__ float fast_tanh(float x) {
    float r;
    asm volatile("v_tanh_f32 %0, %1" : "=v"(r) : "v"(x));
    return r;
}

__device__ __forceinline__ float silu(float x) {
    float s = 0.5f * x;
    float t = fast_tanh(s);
    return fmaf(s, t, s);           // s*(1+tanh(s))
}

__device__ __forceinline__ v8f wmma16(v16h a, v16h b, v8f c) {
    return __builtin_amdgcn_wmma_f32_16x16x32_f16(
        /*neg_a=*/false, a, /*neg_b=*/false, b,
        /*c_mod=*/(short)0, c, /*reuse_a=*/false, /*reuse_b=*/false);
}

// Build one B fragment (K x 16 tile of W^T, i.e. B[k][n] = w[n*ldw + k]).
// B layout (wave32, f16 32x16): lane holds column n = lane&15,
// halves 0..7 hold K = kbase + 8*(lane>=16) + i, halves 8..15 hold K = kbase + 16 + 8*(lane>=16) + (i-8).
__device__ __forceinline__ v16h make_bfrag(const float* __restrict__ w, int ldw,
                                           int n, int kbase, int khi,
                                           int kmax, bool nvalid) {
    v16h f;
#pragma unroll
    for (int i = 0; i < 16; ++i) {
        int k = kbase + (khi >> 1) + ((i < 8) ? i : i + 8);
        float v = 0.f;
        if (nvalid && k < kmax) v = w[n * ldw + k];
        f[i] = (half_t)v;
    }
    return f;
}

// Build A fragments (two k-steps of 32) for the next layer from a 16x64 f32
// LDS row, applying layernorm (mu,rstd,g,be) and SiLU.
// A layout (f16 16x32): lane holds row = lane&15; halves 0..7 -> K=kOff+i,
// halves 8..15 -> K=16+kOff+(i-8), with kOff = 8*(lane>=16).
__device__ __forceinline__ void build_a2(const float* __restrict__ hrow,
                                         const float* __restrict__ g,
                                         const float* __restrict__ be,
                                         int kOff, float mu, float rstd,
                                         v16h a2[2]) {
#pragma unroll
    for (int ks = 0; ks < 2; ++ks) {
        v16h f;
#pragma unroll
        for (int i = 0; i < 16; ++i) {
            int k = ks * 32 + kOff + ((i < 8) ? i : i + 8);
            float v = (hrow[k] - mu) * rstd * g[k] + be[k];
            f[i] = (half_t)silu(v);
        }
        a2[ks] = f;
    }
}

// Row stats, deduplicated across the lane pair (L, L+16): each lane sums 32
// of the 64 columns of its row, then the partials are combined with a
// cross-lane xor-16 shuffle. Halves both ds reads and reduction VALU.
__device__ __forceinline__ void row_stats64_dual(const float* __restrict__ hrowHalf,
                                                 float& mu, float& rstd) {
    float s1 = 0.f, s2 = 0.f;
#pragma unroll
    for (int c = 0; c < 8; ++c) {
        float4 q = reinterpret_cast<const float4*>(hrowHalf)[c];
        s1 += q.x + q.y + q.z + q.w;
        s2 += q.x * q.x + q.y * q.y + q.z * q.z + q.w * q.w;
    }
    s1 += __shfl_xor(s1, 16, 32);
    s2 += __shfl_xor(s2, 16, 32);
    mu = s1 * (1.f / 64.f);
    float var = s2 * (1.f / 64.f) - mu * mu;
    rstd = rsqrtf(var + EPSV);
}

__global__ __launch_bounds__(256)
void mlp_ln_silu_softmax_kernel(
    const float* __restrict__ x,
    const float* __restrict__ w1, const float* __restrict__ b1,
    const float* __restrict__ g1, const float* __restrict__ be1,
    const float* __restrict__ w2, const float* __restrict__ b2,
    const float* __restrict__ g2, const float* __restrict__ be2,
    const float* __restrict__ w3, const float* __restrict__ b3,
    const float* __restrict__ g3, const float* __restrict__ be3,
    float* __restrict__ out, long rows, long numTiles)
{
    __shared__ float sH[8][16 * 64];   // per-wave 16x64 f32 staging
    __shared__ float sP[4 * 64 + 16];  // g1|be1|g2|be2|g3(8)|be3(8)

    const int tid  = threadIdx.x;
    const int lane = tid & 31;
    const int wave = tid >> 5;
    const int col  = lane & 15;
    const int hi   = lane >> 4;        // 0 or 1
    const int kOff = hi * 8;           // A-fragment K striping offset
    const int khi  = hi * 16;          // B-fragment K offset

    if (tid < 64) {
        sP[tid]       = g1[tid];
        sP[64 + tid]  = be1[tid];
        sP[128 + tid] = g2[tid];
        sP[192 + tid] = be2[tid];
    }
    if (tid < 7) {
        sP[256 + tid] = g3[tid];
        sP[264 + tid] = be3[tid];
    }
    __syncthreads();

    float* hbuf = &sH[wave][0];

    // ---- Resident weight fragments (amortized over the tile loop) ----
    v16h B1f[4];
#pragma unroll
    for (int n = 0; n < 4; ++n)
        B1f[n] = make_bfrag(w1, 24, n * 16 + col, 0, khi, 24, true);

    v16h B2f[2][4];
#pragma unroll
    for (int ks = 0; ks < 2; ++ks)
#pragma unroll
        for (int n = 0; n < 4; ++n)
            B2f[ks][n] = make_bfrag(w2, 64, n * 16 + col, ks * 32, khi, 64, true);

    v16h B3f[2];
#pragma unroll
    for (int ks = 0; ks < 2; ++ks)
        B3f[ks] = make_bfrag(w3, 64, col, ks * 32, khi, 64, col < 7);

    float b1r[4], b2r[4];
#pragma unroll
    for (int n = 0; n < 4; ++n) { b1r[n] = b1[n * 16 + col]; b2r[n] = b2[n * 16 + col]; }
    const float b3r = (col < 7) ? b3[col] : 0.f;

    const long waveStride = (long)gridDim.x * 8;

    for (long t = (long)blockIdx.x * 8 + wave; t < numTiles; t += waveStride) {
        const long base = t * 16;

        // ---------- Layer 1: A from global x (K=24 padded to 32) ----------
        long rg = base + (lane & 15);
        if (rg >= rows) rg = rows - 1;           // tail clamp (store is guarded)
        const float* xr = x + rg * 24;

        v16h a1;
        {
            float4 q0 = *reinterpret_cast<const float4*>(xr + kOff);
            float4 q1 = *reinterpret_cast<const float4*>(xr + kOff + 4);
            a1[0] = (half_t)q0.x; a1[1] = (half_t)q0.y;
            a1[2] = (half_t)q0.z; a1[3] = (half_t)q0.w;
            a1[4] = (half_t)q1.x; a1[5] = (half_t)q1.y;
            a1[6] = (half_t)q1.z; a1[7] = (half_t)q1.w;
            if (hi == 0) {  // K = 16..23
                float4 q2 = *reinterpret_cast<const float4*>(xr + 16);
                float4 q3 = *reinterpret_cast<const float4*>(xr + 20);
                a1[8]  = (half_t)q2.x; a1[9]  = (half_t)q2.y;
                a1[10] = (half_t)q2.z; a1[11] = (half_t)q2.w;
                a1[12] = (half_t)q3.x; a1[13] = (half_t)q3.y;
                a1[14] = (half_t)q3.z; a1[15] = (half_t)q3.w;
            } else {        // K = 24..31 -> zero pad
#pragma unroll
                for (int i = 8; i < 16; ++i) a1[i] = (half_t)0.f;
            }
        }

        v8f acc[4];
        const v8f zc = {};
#pragma unroll
        for (int n = 0; n < 4; ++n) acc[n] = wmma16(a1, B1f[n], zc);

        // D layout: lane=col(n*16+col), VGPR r -> row r + 8*hi; add bias, spill to LDS
#pragma unroll
        for (int n = 0; n < 4; ++n)
#pragma unroll
            for (int r = 0; r < 8; ++r)
                hbuf[(r + 8 * hi) * 64 + n * 16 + col] = acc[n][r] + b1r[n];
        lds_fence();

        // ---------- LN + SiLU -> A fragments for layer 2 ----------
        const float* hrow = hbuf + (lane & 15) * 64;
        float mu, rstd;
        row_stats64_dual(hrow + hi * 32, mu, rstd);
        v16h a2[2];
        build_a2(hrow, &sP[0], &sP[64], kOff, mu, rstd, a2);

        // ---------- Layer 2: 64x64 ----------
#pragma unroll
        for (int n = 0; n < 4; ++n) {
            acc[n] = wmma16(a2[0], B2f[0][n], zc);
            acc[n] = wmma16(a2[1], B2f[1][n], acc[n]);
        }
#pragma unroll
        for (int n = 0; n < 4; ++n)
#pragma unroll
            for (int r = 0; r < 8; ++r)
                hbuf[(r + 8 * hi) * 64 + n * 16 + col] = acc[n][r] + b2r[n];
        lds_fence();

        row_stats64_dual(hrow + hi * 32, mu, rstd);
        v16h a3[2];
        build_a2(hrow, &sP[128], &sP[192], kOff, mu, rstd, a3);

        // ---------- Layer 3: 64 -> 7 (N padded to 16) ----------
        v8f acc3 = wmma16(a3[0], B3f[0], zc);
        acc3 = wmma16(a3[1], B3f[1], acc3);
#pragma unroll
        for (int r = 0; r < 8; ++r)
            hbuf[(r + 8 * hi) * 16 + col] = acc3[r] + b3r;
        lds_fence();

        // ---------- LN(7) + softmax(7), one row per lane (lanes 0..15) ----------
        if (lane < 16 && (base + lane) < rows) {
            const float* fr = hbuf + lane * 16;
            float v[7];
            float s1 = 0.f;
#pragma unroll
            for (int j = 0; j < 7; ++j) { v[j] = fr[j]; s1 += v[j]; }
            float m = s1 * (1.f / 7.f);
            float s2 = 0.f;
#pragma unroll
            for (int j = 0; j < 7; ++j) { float d = v[j] - m; s2 += d * d; }
            float rs = rsqrtf(s2 * (1.f / 7.f) + EPSV);
            float y[7], mx = -3.4e38f;
#pragma unroll
            for (int j = 0; j < 7; ++j) {
                y[j] = (v[j] - m) * rs * sP[256 + j] + sP[264 + j];
                mx = fmaxf(mx, y[j]);
            }
            float se = 0.f;
#pragma unroll
            for (int j = 0; j < 7; ++j) { y[j] = __expf(y[j] - mx); se += y[j]; }
            float inv = 1.f / se;
            float* orow = out + (base + lane) * 7;
#pragma unroll
            for (int j = 0; j < 7; ++j) orow[j] = y[j] * inv;
        }
        // next iteration's LDS stores are ordered after these reads (DS in-order)
    }
}

extern "C" void kernel_launch(void* const* d_in, const int* in_sizes, int n_in,
                              void* d_out, int out_size, void* d_ws, size_t ws_size,
                              hipStream_t stream) {
    const float* x   = (const float*)d_in[0];
    const float* w1  = (const float*)d_in[1];
    const float* b1  = (const float*)d_in[2];
    const float* g1  = (const float*)d_in[3];
    const float* be1 = (const float*)d_in[4];
    const float* w2  = (const float*)d_in[5];
    const float* b2  = (const float*)d_in[6];
    const float* g2  = (const float*)d_in[7];
    const float* be2 = (const float*)d_in[8];
    const float* w3  = (const float*)d_in[9];
    const float* b3  = (const float*)d_in[10];
    const float* g3  = (const float*)d_in[11];
    const float* be3 = (const float*)d_in[12];

    long rows = (long)in_sizes[0] / 24;
    long numTiles = (rows + 15) / 16;
    long blocksNeeded = (numTiles + 7) / 8;       // 8 waves (tiles) per block
    int blocks = (int)(blocksNeeded < 4096 ? blocksNeeded : 4096);
    if (blocks < 1) blocks = 1;

    mlp_ln_silu_softmax_kernel<<<blocks, 256, 0, stream>>>(
        x, w1, b1, g1, be1, w2, b2, g2, be2, w3, b3, g3, be3,
        (float*)d_out, rows, numTiles);
}
